// MultiHeadMemoryBank_26499948216634
// MI455X (gfx1250) — matrix operand order repaired and checked by reading, not compiled, not run on hardware
//
#include <hip/hip_runtime.h>
#include <hip/hip_bf16.h>

typedef __attribute__((ext_vector_type(16))) _Float16 v16h;
typedef __attribute__((ext_vector_type(8)))  float    v8f;

#define B_  64
#define H_  8
#define N_  8192
#define D_  64
#define K_  16
#define NEG_INF (-3.402823466e38f)

// ---------------------------------------------------------------------------
// Pass 1: similarity GEMM via WMMA.
// One wave handles a 16x16 tile: 16 keys (8 read + 8 write, pre-scaled by
// beta/||key||) x 16 memory rows (pre-scaled by 1/||row||), K = D = 64 as two
// v_wmma_f32_16x16x32_f16 steps.  C tile == sim directly.
// sim layout: [B][16][N] with keys 0..7 = read heads, 8..15 = write heads,
// so the store address is linear in the C row index -> no divergent branches.
// ---------------------------------------------------------------------------
__global__ __launch_bounds__(256)
void sim_kernel(const float* __restrict__ memory,
                const float* __restrict__ read_keys,
                const float* __restrict__ write_keys,
                const float* __restrict__ beta_r,
                const float* __restrict__ beta_w,
                float* __restrict__ sim)          // [B][16][N]
{
    __shared__ _Float16 kf16[16][D_];

    const int b     = blockIdx.x / (N_ / 128);
    const int chunk = blockIdx.x % (N_ / 128);
    const int tid   = threadIdx.x;

    // ---- prepare 16 normalized+scaled keys in LDS (threads 0..15) ----
    if (tid < 16) {
        const float* kp;
        float beta;
        if (tid < 8) { kp = read_keys  + ((size_t)b * H_ + tid)     * D_; beta = beta_r[b * H_ + tid]; }
        else         { kp = write_keys + ((size_t)b * H_ + (tid-8)) * D_; beta = beta_w[b * H_ + (tid-8)]; }
        float ss = 0.f;
        for (int d = 0; d < D_; ++d) { float v = kp[d]; ss += v * v; }
        const float scale = beta / fmaxf(sqrtf(ss), 1e-12f);
        for (int d = 0; d < D_; ++d) kf16[tid][d] = (_Float16)(kp[d] * scale);
    }
    __syncthreads();

    const int wave = tid >> 5;
    const int lane = tid & 31;
    const int half = lane >> 4;     // 0: lanes 0-15, 1: lanes 16-31
    const int m    = lane & 15;

    // ---- A fragments: lane holds row m, K contiguous [half*16, +16) to match B ----
    v16h a0, a1;
#pragma unroll
    for (int j = 0; j < 16; ++j) {
        const int k = half * 16 + j;   // K index 0..31
        a0[j] = kf16[m][k];
        a1[j] = kf16[m][k + 32];
    }

    // ---- B fragments: lane holds column (memory row) m, dims [half*16, +16) ----
    const int n = chunk * 128 + wave * 16 + m;
    const float* row = memory + ((size_t)b * N_ + n) * D_;
    float vals[32];
    float ss = 0.f;
#pragma unroll
    for (int i = 0; i < 16; ++i) { float v = row[half * 16 + i];      vals[i]      = v; ss += v * v; }
#pragma unroll
    for (int i = 0; i < 16; ++i) { float v = row[32 + half * 16 + i]; vals[16 + i] = v; ss += v * v; }
    ss += __shfl_xor(ss, 16, 32);                      // combine the two half-rows
    const float inv = 1.0f / fmaxf(sqrtf(ss), 1e-12f);

    v16h b0, b1;
#pragma unroll
    for (int i = 0; i < 16; ++i) {
        b0[i] = (_Float16)(vals[i]      * inv);
        b1[i] = (_Float16)(vals[16 + i] * inv);
    }

    v8f c = {};
    c = __builtin_amdgcn_wmma_f32_16x16x32_f16(false, a0, false, b0, (short)0, c, false, false);
    c = __builtin_amdgcn_wmma_f32_16x16x32_f16(false, a1, false, b1, (short)0, c, false, false);

    // ---- C layout: lane column N = m, rows r (+8 for upper lanes) ----
    // Branch-free linear store: key = r + half*8, dst row = b*16 + key.
    float* dst = sim + ((size_t)b * 16 + half * 8) * N_ + n;
#pragma unroll
    for (int r = 0; r < 8; ++r) {
        dst[(size_t)r * N_] = c[r];
    }
}

// ---------------------------------------------------------------------------
// Pass 2: top-16 + softmax per (b,key) row of length N.  slot = b*16+key.
// ---------------------------------------------------------------------------
__global__ __launch_bounds__(256)
void topk_kernel(const float* __restrict__ sim,
                 int* __restrict__ tk_idx, float* __restrict__ tk_w)
{
    __shared__ float s[N_];
    __shared__ float rv[256];
    __shared__ int   ri[256];
    __shared__ float topv[K_];
    __shared__ int   topi[K_];

    const int slot = blockIdx.x;            // [B*16), contiguous with sim layout
    const float* p = sim + (size_t)slot * N_;
    const int tid  = threadIdx.x;

    for (int i = tid; i < N_; i += 256) s[i] = p[i];
    __syncthreads();

    for (int it = 0; it < K_; ++it) {
        float best = NEG_INF; int bi = N_;
        for (int i = tid; i < N_; i += 256) {
            const float v = s[i];
            if (v > best || (v == best && i < bi)) { best = v; bi = i; }
        }
        rv[tid] = best; ri[tid] = bi;
        __syncthreads();
        for (int stp = 128; stp > 0; stp >>= 1) {
            if (tid < stp) {
                const float ov = rv[tid + stp]; const int oi = ri[tid + stp];
                if (ov > rv[tid] || (ov == rv[tid] && oi < ri[tid])) { rv[tid] = ov; ri[tid] = oi; }
            }
            __syncthreads();
        }
        if (tid == 0) { topv[it] = rv[0]; topi[it] = ri[0]; s[ri[0]] = NEG_INF; }
        __syncthreads();
    }

    if (tid == 0) {
        const float mx = topv[0];           // iteration 0 found the global max
        float e[K_]; float sum = 0.f;
        for (int j = 0; j < K_; ++j) { e[j] = __expf(topv[j] - mx); sum += e[j]; }
        const float invs = 1.0f / sum;
        for (int j = 0; j < K_; ++j) {
            tk_idx[slot * K_ + j] = topi[j];
            tk_w [slot * K_ + j] = e[j] * invs;
        }
    }
}

// ---------------------------------------------------------------------------
// Pass 3: sparse read + head projection -> out[:, N, :]
// read slot for (b,h) = b*16 + h.
// ---------------------------------------------------------------------------
__global__ __launch_bounds__(256)
void read_kernel(const float* __restrict__ memory,
                 const int* __restrict__ tk_idx, const float* __restrict__ tk_w,
                 const float* __restrict__ head_w, const float* __restrict__ head_b,
                 float* __restrict__ out)
{
    __shared__ float rph[H_ * D_];          // 512 floats
    const int b   = blockIdx.x;
    const int tid = threadIdx.x;

    for (int t = tid; t < H_ * D_; t += 256) {
        const int h = t >> 6, d = t & 63;
        const int slot = b * 16 + h;        // read slots (keys 0..7)
        float acc = 0.f;
        for (int k = 0; k < K_; ++k) {
            const int n = tk_idx[slot * K_ + k];
            acc += tk_w[slot * K_ + k] * memory[((size_t)b * N_ + n) * D_ + d];
        }
        rph[t] = acc;
    }
    __syncthreads();

    if (tid < D_) {
        const float* wrow = head_w + (size_t)tid * (H_ * D_);
        float acc = head_b[tid];
        for (int j = 0; j < H_ * D_; ++j) acc += wrow[j] * rph[j];
        out[((size_t)b * (N_ + 1) + N_) * D_ + tid] = acc;
    }
}

// ---------------------------------------------------------------------------
// Pass 4: scatter write weights into cw[b][n][h] (unique slots, no atomics)
// write slot for (b,h) = b*16 + 8 + h.
// ---------------------------------------------------------------------------
__global__ void scatter_kernel(const int* __restrict__ tk_idx,
                               const float* __restrict__ tk_w,
                               float* __restrict__ cw)
{
    const int t = blockIdx.x * 256 + threadIdx.x;
    if (t >= B_ * H_ * K_) return;
    const int k = t % K_;
    const int h = (t / K_) % H_;
    const int b = t / (K_ * H_);
    const int slot = b * 16 + 8 + h;        // write slots (keys 8..15)
    const int n = tk_idx[slot * K_ + k];
    cw[((size_t)b * N_ + n) * H_ + h] = tk_w[slot * K_ + k];
}

// ---------------------------------------------------------------------------
// Pass 5: memory update + renorm + decay.  One wave per row (wave32).
// ---------------------------------------------------------------------------
__global__ __launch_bounds__(256)
void update_kernel(const float* __restrict__ memory, const float* __restrict__ cw,
                   const float* __restrict__ erase, const float* __restrict__ add_gate,
                   const float* __restrict__ write_vals, const float* __restrict__ decay_gate,
                   float* __restrict__ out)
{
    const int tid  = threadIdx.x;
    const int wave = tid >> 5;
    const int lane = tid & 31;
    const long row = (long)blockIdx.x * 8 + wave;    // [0, B*N)
    const int b = (int)(row / N_);
    const int n = (int)(row % N_);

    const float* cwp = cw + (size_t)row * H_;
    const int d0 = lane, d1 = lane + 32;
    float ec = 0.f, addv0 = 0.f, addv1 = 0.f;
#pragma unroll
    for (int h = 0; h < H_; ++h) {
        const float w = cwp[h];
        ec += w * erase[b * H_ + h];
        const float wa = w * add_gate[b * H_ + h];
        addv0 += wa * write_vals[((size_t)b * H_ + h) * D_ + d0];
        addv1 += wa * write_vals[((size_t)b * H_ + h) * D_ + d1];
    }
    ec *= (1.0f / H_);

    const float* mrow = memory + (size_t)row * D_;
    float v0 = mrow[d0] * (1.f - ec) + addv0 * (1.0f / H_) + 1e-8f;
    float v1 = mrow[d1] * (1.f - ec) + addv1 * (1.0f / H_) + 1e-8f;

    float ss = v0 * v0 + v1 * v1;
#pragma unroll
    for (int off = 16; off > 0; off >>= 1) ss += __shfl_xor(ss, off, 32);
    float g = 1.0f / fmaxf(sqrtf(ss), 1e-12f);
    g *= 1.0f / (1.0f + __expf(-decay_gate[n]));     // sigmoid(decay)

    float* orow = out + ((size_t)b * (N_ + 1) + n) * D_;
    orow[d0] = v0 * g;
    orow[d1] = v1 * g;
}

// ---------------------------------------------------------------------------
extern "C" void kernel_launch(void* const* d_in, const int* in_sizes, int n_in,
                              void* d_out, int out_size, void* d_ws, size_t ws_size,
                              hipStream_t stream)
{
    (void)in_sizes; (void)n_in; (void)out_size; (void)ws_size;

    const float* memory     = (const float*)d_in[0];
    const float* read_keys  = (const float*)d_in[1];
    const float* write_keys = (const float*)d_in[2];
    const float* write_vals = (const float*)d_in[3];
    const float* erase      = (const float*)d_in[4];
    const float* add_gate   = (const float*)d_in[5];
    const float* beta_r     = (const float*)d_in[6];
    const float* beta_w     = (const float*)d_in[7];
    const float* head_w     = (const float*)d_in[8];
    const float* head_b     = (const float*)d_in[9];
    const float* decay_gate = (const float*)d_in[10];
    float* out = (float*)d_out;

    // workspace layout (floats):
    //   sim    : [B][16][N]   (32 MB) -- keys 0..7 read, 8..15 write
    //   tk_idx : B*16*K ints
    //   tk_w   : B*16*K floats
    //   cw     : aliases the front of sim (dead after top-k), [B][N][H] (16 MB)
    float* sim = (float*)d_ws;
    const size_t simElems = (size_t)B_ * 16 * N_;
    int*   tk_idx = (int*)(sim + simElems);
    float* tk_w   = (float*)(tk_idx + B_ * 16 * K_);
    float* cw     = sim;                     // reuse: sim fully consumed by top-k

    // 1) similarity (WMMA)
    sim_kernel<<<B_ * (N_ / 128), 256, 0, stream>>>(memory, read_keys, write_keys,
                                                    beta_r, beta_w, sim);
    // 2) top-k + softmax for all 16 key rows per batch
    topk_kernel<<<B_ * 16, 256, 0, stream>>>(sim, tk_idx, tk_w);
    // 3) read output row (out[:, N, :])
    read_kernel<<<B_, 256, 0, stream>>>(memory, tk_idx, tk_w, head_w, head_b, out);
    // 4) zero + scatter write coefficients (sim region is dead now)
    hipMemsetAsync(cw, 0, (size_t)B_ * N_ * H_ * sizeof(float), stream);
    scatter_kernel<<<(B_ * H_ * K_ + 255) / 256, 256, 0, stream>>>(tk_idx, tk_w, cw);
    // 5) memory update + renorm + decay (out[:, :N, :])
    update_kernel<<<(B_ * N_) / 8, 256, 0, stream>>>(memory, cw, erase, add_gate,
                                                     write_vals, decay_gate, out);
}
